// GRU_438086664243
// MI455X (gfx1250) — matrix operand order repaired and checked by reading, not compiled, not run on hardware
//
#include <hip/hip_runtime.h>

typedef __attribute__((ext_vector_type(16))) _Float16 v16h;
typedef __attribute__((ext_vector_type(8)))  float    v8f;

#define NT_STEPS 1024
#define NBATCH   64
#define NIN      256
#define NHID     512
#define NOUT     128
#define NROWS    (NT_STEPS * NBATCH)   /* 65536 */
#define GWGS     32                    /* persistent recurrence workgroups */
#define KT_H     (NHID / 32)           /* 16 K-tiles over hidden dim */
#define KT_X     (NIN  / 32)           /* 8  K-tiles over input dim  */

static __device__ __forceinline__ v8f wmma16(v16h a, v16h b, v8f c) {
  // D = A(16x32 f16) * B(32x16 f16) + C(16x16 f32)
  return __builtin_amdgcn_wmma_f32_16x16x32_f16(false, a, false, b, (short)0, c,
                                                false, false);
}

// Build an A fragment (16x32 f16 tile) from a row-major f16 matrix.
// Per-lane layout (ISA 7.12.2): lane L (row m = 16*mt + L%16, g = L/16) holds
// halves e=0..7  at K = kbase + 8g + e
// halves e=8..15 at K = kbase + 16 + 8g + (e-8)
static __device__ __forceinline__ v16h load_a_f16(const _Float16* __restrict__ base,
                                                  int ld, int m, int kbase, int g) {
  const uint4* p = (const uint4*)(base + (size_t)m * ld + kbase + g * 8);
  union { uint4 u[2]; v16h h; } u;
  u.u[0] = p[0];   // K = kbase + 8g .. +7
  u.u[1] = p[2];   // +32B = +16 halves
  return u.h;
}

// ---------------------------------------------------------------- prep kernels
__global__ void k_f32_to_f16(const float* __restrict__ src,
                             _Float16* __restrict__ dst, int n) {
  for (int i = blockIdx.x * blockDim.x + threadIdx.x; i < n;
       i += gridDim.x * blockDim.x)
    dst[i] = (_Float16)src[i];
}

// Pre-swizzle a [K x N] f32 matrix into WMMA B-fragment order (f16).
// dst[((kt*NT + nt)*32 + L)*16 + e] = src[(32*kt + 16*(e/8) + 8*(L/16) + e%8) , 16*nt + L%16]
__global__ void k_swizzleB(const float* __restrict__ src,
                           _Float16* __restrict__ dst,
                           int K, int N, int transposed) {
  int NT = N >> 4;
  int total = K * N;
  for (int o = blockIdx.x * blockDim.x + threadIdx.x; o < total;
       o += gridDim.x * blockDim.x) {
    int e    = o & 15;
    int L    = (o >> 4) & 31;
    int tile = o >> 9;
    int kt   = tile / NT;
    int nt   = tile - kt * NT;
    int k    = kt * 32 + ((e >> 3) << 4) + ((L >> 4) << 3) + (e & 7);
    int n    = nt * 16 + (L & 15);
    float v  = transposed ? src[(size_t)n * K + k] : src[(size_t)k * N + n];
    dst[o] = (_Float16)v;
  }
}

__global__ void k_zero32(unsigned* __restrict__ dst, int n) {
  for (int i = blockIdx.x * blockDim.x + threadIdx.x; i < n;
       i += gridDim.x * blockDim.x)
    dst[i] = 0u;
}

// ---------------------------------------- fused input projections (all 3 gates)
// Each wave owns one M-tile x two N-tiles, computes z/r/h gates together:
// one A fragment feeds 6 wmma. M=65536, N=512, K=256. Outputs stored f16.
__global__ __launch_bounds__(256) void k_xproj3(const _Float16* __restrict__ X,
                                                const _Float16* __restrict__ WzS,
                                                const _Float16* __restrict__ WrS,
                                                const _Float16* __restrict__ WhS,
                                                const float* __restrict__ bz,
                                                const float* __restrict__ br,
                                                const float* __restrict__ bh,
                                                _Float16* __restrict__ Pz,
                                                _Float16* __restrict__ Pr,
                                                _Float16* __restrict__ Ph) {
  const int lane = threadIdx.x & 31;
  const int w    = threadIdx.x >> 5;
  const int tileid = blockIdx.x * 8 + w;     // 4096 mt x 16 nt-pairs = 65536
  const int mt  = tileid >> 4;
  const int ntp = tileid & 15;               // pair of N-tiles
  const int g   = lane >> 4;
  const int m0  = mt * 16 + (lane & 15);

  const _Float16* Wp[3] = {WzS, WrS, WhS};
  v8f acc[3][2] = {};
  for (int kt = 0; kt < KT_X; ++kt) {
    v16h a = load_a_f16(X, NIN, m0, kt * 32, g);
#pragma unroll
    for (int gate = 0; gate < 3; ++gate) {
#pragma unroll
      for (int j = 0; j < 2; ++j) {
        int nt = ntp * 2 + j;
        v16h b = *(const v16h*)(Wp[gate] +
                                ((size_t)(kt * 32 + nt) * 32 + lane) * 16);
        acc[gate][j] = wmma16(a, b, acc[gate][j]);
      }
    }
  }
  const float* Bp[3] = {bz, br, bh};
  _Float16* Op[3] = {Pz, Pr, Ph};
#pragma unroll
  for (int gate = 0; gate < 3; ++gate) {
#pragma unroll
    for (int j = 0; j < 2; ++j) {
      int col = (ntp * 2 + j) * 16 + (lane & 15);
      float bv = Bp[gate][col];
#pragma unroll
      for (int r = 0; r < 8; ++r) {
        int m = mt * 16 + g * 8 + r;
        Op[gate][(size_t)m * NHID + col] = (_Float16)(acc[gate][j][r] + bv);
      }
    }
  }
}

// ------------------------------------------------------- persistent GRU scan
static __device__ __forceinline__ void grid_sync(unsigned* cnt, unsigned& epoch) {
  __syncthreads();
  if (threadIdx.x == 0) {
    __threadfence();
    epoch += 1u;
    atomicAdd(cnt, 1u);
    const unsigned target = epoch * GWGS;
    volatile unsigned* vc = (volatile unsigned*)cnt;
    while (*vc < target) __builtin_amdgcn_s_sleep(1);
    __threadfence();
  }
  __syncthreads();
}

__global__ __launch_bounds__(256) void k_gru(const _Float16* __restrict__ Pz,
                                             const _Float16* __restrict__ Pr,
                                             const _Float16* __restrict__ Ph,
                                             const _Float16* __restrict__ WhzS,
                                             const _Float16* __restrict__ WhrS,
                                             const _Float16* __restrict__ WhhS,
                                             _Float16* __restrict__ Hf16,
                                             _Float16* __restrict__ RHf16,
                                             float*    __restrict__ outH,
                                             unsigned* __restrict__ sync) {
  // Recurrent weight slices for this WG's 16 columns: 3 gates x 16 ktiles x 512 halves
  __shared__ __attribute__((aligned(32))) _Float16 s_w[3][KT_H][512]; // 48 KB
  __shared__ float s_red[4][32][8];                                   // 4 KB

  const int tid  = threadIdx.x;
  const int lane = tid & 31;
  const int w    = tid >> 5;            // 0..7 waves
  const int nt   = blockIdx.x;          // global N-tile (0..31)
  const int colbase = nt * 16;
  const int g   = lane >> 4;
  const int col = lane & 15;

  // stage weight slices into LDS (uint4 chunks)
  {
    _Float16* sflat = &s_w[0][0][0];
    for (int j = tid; j < 3 * KT_H * 512 / 8; j += 256) {
      int hidx = j * 8;
      int gate = hidx >> 13;            // /8192
      int rem  = hidx & 8191;
      int kt   = rem >> 9;
      int off  = rem & 511;
      const _Float16* src = (gate == 0) ? WhzS : ((gate == 1) ? WhrS : WhhS);
      *(uint4*)(sflat + hidx) =
          *(const uint4*)(src + (size_t)(kt * 32 + nt) * 512 + off);
    }
  }
  __syncthreads();

  const int mt = w & 3;                 // M-tile this wave owns
  const int gatep = w >> 2;             // 0: z-gate path, 1: r-gate path
  const int m0 = mt * 16 + (lane & 15);

  unsigned epoch = 0;
  for (int t = 0; t < NT_STEPS; ++t) {
    const size_t prow = (size_t)t * NBATCH;

    // prefetch next timestep's projections into cache (global_prefetch_b8)
    if (t + 1 < NT_STEPS) {
      size_t nrow = (prow + NBATCH + mt * 16 + g * 8) * NHID + colbase;
      __builtin_prefetch(Pz + nrow, 0, 1);
      __builtin_prefetch(Pr + nrow, 0, 1);
      __builtin_prefetch(Ph + nrow, 0, 1);
    }

    // -------- phase 1: Z (waves 0-3) / R + R*H (waves 4-7), own 16 cols
    v8f acc = {};
#pragma unroll
    for (int kt = 0; kt < KT_H; ++kt) {
      v16h a = load_a_f16(Hf16, NHID, m0, kt * 32, g);
      v16h b = *(const v16h*)&s_w[gatep][kt][lane * 16];
      acc = wmma16(a, b, acc);
    }
    const _Float16* P = gatep ? Pr : Pz;
    float gv[8];
#pragma unroll
    for (int r = 0; r < 8; ++r) {
      int m = mt * 16 + g * 8 + r;
      float x = (float)P[(prow + m) * NHID + colbase + col];
      gv[r] = 1.0f / (1.0f + __expf(-(acc[r] + x)));
    }
    if (gatep == 1) {                   // R waves: publish R .* H
#pragma unroll
      for (int r = 0; r < 8; ++r) {
        int m = mt * 16 + g * 8 + r;
        size_t idx = (size_t)m * NHID + colbase + col;
        RHf16[idx] = (_Float16)(gv[r] * (float)Hf16[idx]);
      }
    }
    grid_sync(sync, epoch);             // full R*H now visible

    // -------- phase 2: Htilda = tanh(xh + (R.*H) @ Whh), K split 8/8 per pair
    v8f acc2 = {};
    const int ktlo = (w < 4) ? 0 : 8;
#pragma unroll
    for (int kk = 0; kk < 8; ++kk) {
      int kt = ktlo + kk;
      v16h a = load_a_f16(RHf16, NHID, m0, kt * 32, g);
      v16h b = *(const v16h*)&s_w[2][kt][lane * 16];
      acc2 = wmma16(a, b, acc2);
    }
    if (w >= 4) {
#pragma unroll
      for (int r = 0; r < 8; ++r) s_red[mt][lane][r] = acc2[r];
    }
    __syncthreads();

    // -------- phase 3: combine (waves 0-3 hold Z)
    if (w < 4) {
#pragma unroll
      for (int r = 0; r < 8; ++r) {
        int m = mt * 16 + g * 8 + r;
        size_t idx = (size_t)m * NHID + colbase + col;
        float s = acc2[r] + s_red[mt][lane][r] +
                  (float)Ph[(prow + m) * NHID + colbase + col];
        float ht = tanhf(s);
        float ho = (float)Hf16[idx];
        float z  = gv[r];
        float hn = z * ho + (1.0f - z) * ht;
        outH[(prow + m) * NHID + colbase + col] = hn;
        Hf16[idx] = (_Float16)hn;
      }
    }
    grid_sync(sync, epoch);             // H fully updated for next step
  }
}

// ------------------------------------------------ readout: relu(H) @ fcW^T + b
// Each wave owns one M-tile and ALL 8 N-tiles: one A build feeds 8 wmma.
__global__ __launch_bounds__(256) void k_fc(const float*    __restrict__ outF,
                                            const _Float16* __restrict__ FcS,
                                            const float*    __restrict__ fcb,
                                            float*          __restrict__ R) {
  const int lane = threadIdx.x & 31;
  const int w    = threadIdx.x >> 5;
  const int mt   = blockIdx.x * 8 + w;        // 4096 M-tiles
  const int g    = lane >> 4;
  const int m0   = mt * 16 + (lane & 15);

  v8f acc[8] = {};
  for (int kt = 0; kt < KT_H; ++kt) {
    const float4* p4 = (const float4*)(outF + (size_t)m0 * NHID + kt * 32 + g * 8);
    float4 f0 = p4[0], f1 = p4[1], f2 = p4[4], f3 = p4[5];
    float tmp[16] = {f0.x, f0.y, f0.z, f0.w, f1.x, f1.y, f1.z, f1.w,
                     f2.x, f2.y, f2.z, f2.w, f3.x, f3.y, f3.z, f3.w};
    union { v16h h; _Float16 e[16]; } ua;
#pragma unroll
    for (int i = 0; i < 16; ++i) ua.e[i] = (_Float16)fmaxf(tmp[i], 0.0f);
#pragma unroll
    for (int nt = 0; nt < 8; ++nt) {
      v16h b = *(const v16h*)(FcS + ((size_t)(kt * 8 + nt) * 32 + lane) * 16);
      acc[nt] = wmma16(ua.h, b, acc[nt]);
    }
  }
#pragma unroll
  for (int nt = 0; nt < 8; ++nt) {
    const int col = nt * 16 + (lane & 15);
    const float bv = fcb[col];
#pragma unroll
    for (int r = 0; r < 8; ++r) {
      int m = mt * 16 + g * 8 + r;
      R[(size_t)m * NOUT + col] = acc[nt][r] + bv;
    }
  }
}

// --------------------------------------------------------------------- launch
extern "C" void kernel_launch(void* const* d_in, const int* in_sizes, int n_in,
                              void* d_out, int out_size, void* d_ws, size_t ws_size,
                              hipStream_t stream) {
  (void)in_sizes; (void)n_in; (void)out_size; (void)ws_size;
  const float* inputs = (const float*)d_in[0];
  const float* W_xz = (const float*)d_in[1];
  const float* W_hz = (const float*)d_in[2];
  const float* b_z  = (const float*)d_in[3];
  const float* W_xr = (const float*)d_in[4];
  const float* W_hr = (const float*)d_in[5];
  const float* b_r  = (const float*)d_in[6];
  const float* W_xh = (const float*)d_in[7];
  const float* W_hh = (const float*)d_in[8];
  const float* b_h  = (const float*)d_in[9];
  const float* fc_w = (const float*)d_in[10];
  const float* fc_b = (const float*)d_in[11];

  char* ws = (char*)d_ws;
  size_t off = 0;
  auto carve = [&](size_t bytes) -> char* {
    char* p = ws + off;
    off += (bytes + 255) & ~(size_t)255;
    return p;
  };
  _Float16* Xf16 = (_Float16*)carve((size_t)NROWS * NIN * 2);
  _Float16* Pz   = (_Float16*)carve((size_t)NROWS * NHID * 2);
  _Float16* Pr   = (_Float16*)carve((size_t)NROWS * NHID * 2);
  _Float16* Ph   = (_Float16*)carve((size_t)NROWS * NHID * 2);
  _Float16* WhzS = (_Float16*)carve((size_t)NHID * NHID * 2);
  _Float16* WhrS = (_Float16*)carve((size_t)NHID * NHID * 2);
  _Float16* WhhS = (_Float16*)carve((size_t)NHID * NHID * 2);
  _Float16* WxzS = (_Float16*)carve((size_t)NIN * NHID * 2);
  _Float16* WxrS = (_Float16*)carve((size_t)NIN * NHID * 2);
  _Float16* WxhS = (_Float16*)carve((size_t)NIN * NHID * 2);
  _Float16* FcS  = (_Float16*)carve((size_t)NHID * NOUT * 2);
  _Float16* Hf16 = (_Float16*)carve((size_t)NBATCH * NHID * 2);
  _Float16* RHf16= (_Float16*)carve((size_t)NBATCH * NHID * 2);
  unsigned* syncc= (unsigned*)carve(256);

  float* outH = (float*)d_out;                       // [T,B,NH]
  float* outR = outH + (size_t)NROWS * NHID;         // [T,B,NO]

  // prep
  k_f32_to_f16<<<2048, 256, 0, stream>>>(inputs, Xf16, NROWS * NIN);
  k_swizzleB<<<512, 256, 0, stream>>>(W_hz, WhzS, NHID, NHID, 0);
  k_swizzleB<<<512, 256, 0, stream>>>(W_hr, WhrS, NHID, NHID, 0);
  k_swizzleB<<<512, 256, 0, stream>>>(W_hh, WhhS, NHID, NHID, 0);
  k_swizzleB<<<256, 256, 0, stream>>>(W_xz, WxzS, NIN, NHID, 0);
  k_swizzleB<<<256, 256, 0, stream>>>(W_xr, WxrS, NIN, NHID, 0);
  k_swizzleB<<<256, 256, 0, stream>>>(W_xh, WxhS, NIN, NHID, 0);
  k_swizzleB<<<128, 256, 0, stream>>>(fc_w, FcS, NHID, NOUT, 1);
  k_zero32<<<64, 256, 0, stream>>>((unsigned*)Hf16, NBATCH * NHID / 2);
  k_zero32<<<1, 64, 0, stream>>>(syncc, 64);

  // fused input projections (f16 out), all three gates in one pass
  k_xproj3<<<8192, 256, 0, stream>>>(Xf16, WxzS, WxrS, WxhS,
                                     b_z, b_r, b_h, Pz, Pr, Ph);

  // persistent recurrence
  k_gru<<<GWGS, 256, 0, stream>>>(Pz, Pr, Ph, WhzS, WhrS, WhhS,
                                  Hf16, RHf16, outH, syncc);

  // per-timestep readout
  k_fc<<<512, 256, 0, stream>>>(outH, FcS, fc_b, outR);
}
